// LoFTREncoderLayer_747324309653
// MI455X (gfx1250) — matrix operand order repaired and checked by reading, not compiled, not run on hardware
//
#include <hip/hip_runtime.h>
#include <hip/hip_bf16.h>

#define D_MODEL 256
#define NHEAD   8
#define DHEAD   32
#define BSZ     8
#define SEQ     4096

typedef unsigned short bfbits;
typedef __bf16 bf16x16 __attribute__((ext_vector_type(16)));
typedef float  v8f     __attribute__((ext_vector_type(8)));
typedef int    v4i     __attribute__((vector_size(16)));   // matches builtin param type

union FragU { uint4 q[2]; bf16x16 v; };

__device__ __forceinline__ bfbits f2bf(float f) {
  unsigned u = __float_as_uint(f);
  return (bfbits)((u + 0x7FFFu + ((u >> 16) & 1u)) >> 16);  // RNE
}
__device__ __forceinline__ float bf2f(bfbits b) {
  return __uint_as_float(((unsigned)b) << 16);
}

// ---------------- CDNA5 async global->LDS staging (guarded; fallback = LDS store)
#if defined(__has_builtin)
#if __has_builtin(__builtin_amdgcn_global_load_async_to_lds_b128)
#define HAS_ASYNC_LDS 1
#endif
#if __has_builtin(__builtin_amdgcn_s_wait_asynccnt)
#define HAS_WAIT_ASYNC 1
#endif
#endif

__device__ __forceinline__ void async_copy16(const bfbits* g, bfbits* l) {
#ifdef HAS_ASYNC_LDS
  __builtin_amdgcn_global_load_async_to_lds_b128(
      (__attribute__((address_space(1))) v4i*)(bfbits*)g,
      (__attribute__((address_space(3))) v4i*)l, 0, 0);
#else
  *(uint4*)l = *(const uint4*)g;
#endif
}

template <int N>
__device__ __forceinline__ void async_wait() {
#ifdef HAS_ASYNC_LDS
#ifdef HAS_WAIT_ASYNC
  __builtin_amdgcn_s_wait_asynccnt(N);
#else
  asm volatile("s_wait_asynccnt %0" ::"i"(N) : "memory");
#endif
#endif
}

// A-fragment: 16x32 bf16 (MxK). lane m=lane&15, hf=lane>>4 holds
// K = {16*hf..16*hf+15} (VGPR0..7).
__device__ __forceinline__ bf16x16 load_frag_a(const bfbits* A, int ld, int m0, int k0) {
  const int lane = threadIdx.x & 31;
  const int m  = m0 + (lane & 15);
  const int hf = lane >> 4;
  const bfbits* p = A + (size_t)m * ld + k0 + hf * 16;
  FragU f;
  f.q[0] = *(const uint4*)(p);
  f.q[1] = *(const uint4*)(p + 8);
  return f.v;
}

// B-fragment from global: 32x16 (KxN), B = W^T with W[N,K] row-major.
__device__ __forceinline__ bf16x16 load_frag_b(const bfbits* W, int ld, int n0, int k0) {
  const int lane = threadIdx.x & 31;
  const int n  = n0 + (lane & 15);
  const int hf = lane >> 4;
  const bfbits* p = W + (size_t)n * ld + k0 + hf * 16;
  FragU f;
  f.q[0] = *(const uint4*)(p);
  f.q[1] = *(const uint4*)(p + 8);
  return f.v;
}

// B-fragment from LDS tile sB[n][32] (row-major, 32 K per row).
__device__ __forceinline__ bf16x16 load_frag_b_lds(const bfbits* sB, int n0) {
  const int lane = threadIdx.x & 31;
  const int n  = n0 + (lane & 15);
  const int hf = lane >> 4;
  const bfbits* p = sB + n * 32 + hf * 16;
  FragU f;
  f.q[0] = *(const uint4*)(p);
  f.q[1] = *(const uint4*)(p + 8);
  return f.v;
}

// ---------------------------------------------------------------- fp32 -> bf16
__global__ __launch_bounds__(256) void cvt_kernel(const float* __restrict__ s,
                                                  bfbits* __restrict__ d, int n4) {
  int i = blockIdx.x * blockDim.x + threadIdx.x;
  if (i < n4) {
    float4 f = ((const float4*)s)[i];
    ushort4 o;
    o.x = f2bf(f.x); o.y = f2bf(f.y); o.z = f2bf(f.z); o.w = f2bf(f.w);
    ((ushort4*)d)[i] = o;
  }
}

// ------------------------------------------------- generic Y = A * W^T (bf16 in/out)
// EPI: 1 = elu(v)+1, 2 = v*scale, 3 = relu(v)
// block = 8 waves; block tile 128(M) x 16*NT(N); wave tile 16(M) x 16*NT(N)
// B tile (32K x 16*NT) double-buffered in LDS via async copy; in-order ASYNCcnt
// completion lets s_wait_asynccnt<=4 release the older step while the newer streams.
template <int EPI, int NT>
__global__ __launch_bounds__(256) void gemm_bf16(const bfbits* __restrict__ A,
                                                 const bfbits* __restrict__ W,
                                                 bfbits* __restrict__ Y,
                                                 int M, int N, int K, float scale) {
  __shared__ bfbits sB[2][NT * 16 * 32];
  const int tid    = threadIdx.x;
  const int w      = tid >> 5;
  const int lane   = tid & 31;
  const int n_lane = lane & 15;
  const int hf     = lane >> 4;
  const int m0 = blockIdx.y * 128 + w * 16;
  const int nb = blockIdx.x * (NT * 16);
  v8f acc[NT] = {};

  auto stage = [&](int k0, int buf) {
#pragma unroll
    for (int j = 0; j < NT / 4; ++j) {
      int c = tid + 256 * j;           // 16B chunk id
      int n_loc = c >> 2;
      int seg   = c & 3;
      async_copy16(W + (size_t)(nb + n_loc) * K + k0 + seg * 8,
                   &sB[buf][n_loc * 32 + seg * 8]);
    }
  };

  const int steps = K / 32;
  stage(0, 0);
  for (int i = 0; i < steps; ++i) {
    if (i + 1 < steps) {
      stage((i + 1) * 32, (i + 1) & 1);  // overlap next copy with this compute
      async_wait<NT / 4>();              // older step's copies complete (in-order)
    } else {
      async_wait<0>();
    }
    __syncthreads();
    bf16x16 a = load_frag_a(A, K, m0, i * 32);
    const bfbits* sb = sB[i & 1];
#pragma unroll
    for (int t = 0; t < NT; ++t) {
      bf16x16 b = load_frag_b_lds(sb, 16 * t);
      acc[t] = __builtin_amdgcn_wmma_f32_16x16x32_bf16(false, a, false, b,
                                                       (short)0, acc[t], false, false);
    }
    __syncthreads();
  }
#pragma unroll
  for (int t = 0; t < NT; ++t) {
    const int n = nb + 16 * t + n_lane;
#pragma unroll
    for (int r = 0; r < 8; ++r) {
      const int m = m0 + r + 8 * hf;
      float v = acc[t][r];
      if (EPI == 1)      v = (v > 0.f) ? (v + 1.f) : __expf(v);   // elu(v)+1
      else if (EPI == 2) v = v * scale;
      else if (EPI == 3) v = fmaxf(v, 0.f);
      Y[(size_t)m * N + n] = f2bf(v);
    }
  }
}

// ---------------------------------------- per (b,h): KVt[e][d] = sum_s K[s,d]*V[s,e]
__global__ __launch_bounds__(256) void kv_reduce(const bfbits* __restrict__ Kp,
                                                 const bfbits* __restrict__ Vs,
                                                 bfbits* __restrict__ KVt,
                                                 float* __restrict__ Ksum, int S) {
  __shared__ float sK[64][33];
  __shared__ float sV[64][33];
  const int tid = threadIdx.x;
  const int bh  = blockIdx.x;
  const int b   = bh >> 3, h = bh & 7;
  const int d   = tid >> 3;
  const int e0  = (tid & 7) * 4;
  float acc[4] = {0.f, 0.f, 0.f, 0.f};
  float ks = 0.f;
  const size_t base = (size_t)b * S * 256 + h * 32;
  for (int s0 = 0; s0 < S; s0 += 64) {
#pragma unroll
    for (int j = 0; j < 8; ++j) {
      int li = tid + 256 * j;
      int s = li >> 5, dd = li & 31;
      size_t g = base + (size_t)(s0 + s) * 256 + dd;
      sK[s][dd] = bf2f(Kp[g]);
      sV[s][dd] = bf2f(Vs[g]);
    }
    __syncthreads();
#pragma unroll 4
    for (int s = 0; s < 64; ++s) {
      float kk = sK[s][d];
      acc[0] += kk * sV[s][e0 + 0];
      acc[1] += kk * sV[s][e0 + 1];
      acc[2] += kk * sV[s][e0 + 2];
      acc[3] += kk * sV[s][e0 + 3];
    }
    if (tid < 32) {
      for (int s = 0; s < 64; ++s) ks += sK[s][tid];
    }
    __syncthreads();
  }
#pragma unroll
  for (int i = 0; i < 4; ++i)   // transposed so attend's B-frag loads are contiguous
    KVt[((size_t)bh * 32 + (e0 + i)) * 32 + d] = f2bf(acc[i]);
  if (tid < 32) Ksum[(size_t)bh * 32 + tid] = ks;
}

// ------------------- attended = (Q' @ KV) * S / (Q'.Ksum + eps), per (b,h)
__global__ __launch_bounds__(256) void attend(const bfbits* __restrict__ Qp,
                                              const bfbits* __restrict__ KVt,
                                              const float* __restrict__ Ksum,
                                              bfbits* __restrict__ Att, int L, int S) {
  __shared__ float sscale[128];
  __shared__ float sks[32];
  const int tid = threadIdx.x;
  const int bh  = blockIdx.y;
  const int b   = bh >> 3, h = bh & 7;
  const int l0  = blockIdx.x * 128;
  if (tid < 32) sks[tid] = Ksum[(size_t)bh * 32 + tid];
  __syncthreads();
  const bfbits* Qbase = Qp + (size_t)b * L * 256 + h * 32;
  if (tid < 128) {
    const bfbits* q = Qbase + (size_t)(l0 + tid) * 256;
    float dot = 0.f;
#pragma unroll
    for (int dd = 0; dd < 32; ++dd) dot += bf2f(q[dd]) * sks[dd];
    sscale[tid] = (float)S / (dot + 1e-6f);
  }
  __syncthreads();
  const int w      = tid >> 5;
  const int lane   = tid & 31;
  const int n_lane = lane & 15;
  const int hf     = lane >> 4;
  const int m0 = l0 + w * 16;
  bf16x16 a  = load_frag_a(Qbase, 256, m0, 0);
  const bfbits* kvb = KVt + (size_t)bh * 1024;
  bf16x16 fb0 = load_frag_b(kvb, 32, 0, 0);
  bf16x16 fb1 = load_frag_b(kvb, 32, 16, 0);
  v8f c0 = {}; v8f c1 = {};
  c0 = __builtin_amdgcn_wmma_f32_16x16x32_bf16(false, a, false, fb0, (short)0, c0, false, false);
  c1 = __builtin_amdgcn_wmma_f32_16x16x32_bf16(false, a, false, fb1, (short)0, c1, false, false);
#pragma unroll
  for (int r = 0; r < 8; ++r) {
    int ml = w * 16 + r + 8 * hf;
    float sc = sscale[ml];
    size_t row = (size_t)b * L * 256 + (size_t)(l0 + ml) * 256 + h * 32;
    Att[row + n_lane]      = f2bf(c0[r] * sc);
    Att[row + 16 + n_lane] = f2bf(c1[r] * sc);
  }
}

// --------- Out = R + LayerNorm(A @ W^T) * gamma + beta  (N fixed = 256)
// block = 16 rows x 256 cols (8 waves, wave = 16x32) so LN is block-local.
__global__ __launch_bounds__(256) void gemm_ln(const bfbits* __restrict__ A,
                                               const bfbits* __restrict__ W,
                                               const float* __restrict__ R,
                                               const float* __restrict__ gamma,
                                               const float* __restrict__ beta,
                                               float* __restrict__ Out,
                                               bfbits* __restrict__ OutB, int K) {
  __shared__ float sm[16][264];
  __shared__ float ps[16][17];
  __shared__ float ps2[16][17];
  __shared__ float mu_s[16], rs_s[16];
  const int tid    = threadIdx.x;
  const int w      = tid >> 5;
  const int lane   = tid & 31;
  const int n_lane = lane & 15;
  const int hf     = lane >> 4;
  const int m0 = blockIdx.x * 16;
  const int n0 = w * 32;
  v8f acc0 = {}; v8f acc1 = {};
  for (int k0 = 0; k0 < K; k0 += 32) {
    bf16x16 a   = load_frag_a(A, K, m0, k0);
    bf16x16 fb0 = load_frag_b(W, K, n0, k0);
    bf16x16 fb1 = load_frag_b(W, K, n0 + 16, k0);
    acc0 = __builtin_amdgcn_wmma_f32_16x16x32_bf16(false, a, false, fb0, (short)0, acc0, false, false);
    acc1 = __builtin_amdgcn_wmma_f32_16x16x32_bf16(false, a, false, fb1, (short)0, acc1, false, false);
  }
#pragma unroll
  for (int r = 0; r < 8; ++r) {
    sm[r + 8 * hf][n0 + n_lane]      = acc0[r];
    sm[r + 8 * hf][n0 + 16 + n_lane] = acc1[r];
  }
  __syncthreads();
  {  // parallel LN statistics: 16 threads per row
    const int r = tid >> 4, ii = tid & 15;
    float s = 0.f, s2 = 0.f;
#pragma unroll
    for (int c = ii; c < 256; c += 16) { float v = sm[r][c]; s += v; s2 += v * v; }
    ps[r][ii] = s; ps2[r][ii] = s2;
  }
  __syncthreads();
  if (tid < 16) {
    float s = 0.f, s2 = 0.f;
#pragma unroll
    for (int i = 0; i < 16; ++i) { s += ps[tid][i]; s2 += ps2[tid][i]; }
    float mu = s * (1.0f / 256.0f);
    float var = s2 * (1.0f / 256.0f) - mu * mu;
    mu_s[tid] = mu;
    rs_s[tid] = rsqrtf(var + 1e-5f);
  }
  __syncthreads();
  const int c = tid;
  float g = gamma[c], be = beta[c];
#pragma unroll
  for (int r = 0; r < 16; ++r) {
    size_t idx = (size_t)(m0 + r) * 256 + c;
    float val = R[idx] + (sm[r][c] - mu_s[r]) * rs_s[r] * g + be;
    Out[idx] = val;
    if (OutB) OutB[idx] = f2bf(val);
  }
}

extern "C" void kernel_launch(void* const* d_in, const int* in_sizes, int n_in,
                              void* d_out, int out_size, void* d_ws, size_t ws_size,
                              hipStream_t stream) {
  const float* x   = (const float*)d_in[0];
  const float* src = (const float*)d_in[1];
  const float* wq  = (const float*)d_in[2];
  const float* wk  = (const float*)d_in[3];
  const float* wv  = (const float*)d_in[4];
  const float* wm  = (const float*)d_in[5];
  const float* w1  = (const float*)d_in[6];
  const float* w2  = (const float*)d_in[7];
  const float* g1  = (const float*)d_in[8];
  const float* b1  = (const float*)d_in[9];
  const float* g2  = (const float*)d_in[10];
  const float* b2  = (const float*)d_in[11];

  const int M = BSZ * SEQ;            // 32768 rows
  const size_t MB = 1ull << 20;
  char* ws = (char*)d_ws;
  // buffer reuse plan (lifetimes verified against launch order):
  bfbits* xb  = (bfbits*)(ws + 0  * MB);   // 16MB; dead after Q proj
  bfbits* sb  = (bfbits*)(ws + 16 * MB);   // 16MB; dead after V proj
  bfbits* Qp  = (bfbits*)(ws + 32 * MB);   // 16MB; dead after attend
  bfbits* Kp  = (bfbits*)(ws + 48 * MB);   // 16MB; dead after kv_reduce
  bfbits* Vsb = (bfbits*)(ws + 64 * MB);   // 16MB; dead after kv_reduce
  bfbits* Att = (bfbits*)(ws + 0  * MB);   // reuse xb
  bfbits* x1b = (bfbits*)(ws + 16 * MB);   // reuse sb
  bfbits* Hb  = (bfbits*)(ws + 32 * MB);   // 32MB; reuse Qp+Kp
  float*  x1  = (float*) (ws + 64 * MB);   // 32MB; reuse Vsb + fresh
  char* wsw = ws + 96 * MB;
  bfbits* wqb = (bfbits*)(wsw + 0   * 1024);
  bfbits* wkb = (bfbits*)(wsw + 128 * 1024);
  bfbits* wvb = (bfbits*)(wsw + 256 * 1024);
  bfbits* wmb = (bfbits*)(wsw + 384 * 1024);
  bfbits* w1b = (bfbits*)(wsw + 512 * 1024);
  bfbits* w2b = (bfbits*)(wsw + 768 * 1024);
  bfbits* KVt = (bfbits*)(wsw + 1024 * 1024);
  float* Ksum = (float*) (wsw + 1024 * 1024 + 128 * 1024);

  auto cvt = [&](const float* s, bfbits* dptr, int cnt) {
    int n4 = cnt >> 2;
    cvt_kernel<<<(n4 + 255) / 256, 256, 0, stream>>>(s, dptr, n4);
  };
  cvt(x,   xb,  M * 256);
  cvt(src, sb,  M * 256);
  cvt(wq,  wqb, 256 * 256);
  cvt(wk,  wkb, 256 * 256);
  cvt(wv,  wvb, 256 * 256);
  cvt(wm,  wmb, 256 * 256);
  cvt(w1,  w1b, 512 * 256);
  cvt(w2,  w2b, 256 * 512);

  // Q' = elu(x@wq^T)+1 ; K' = elu(src@wk^T)+1 ; Vs = (src@wv^T)/S
  gemm_bf16<1, 16><<<dim3(1, M / 128), 256, 0, stream>>>(xb, wqb, Qp, M, 256, 256, 1.f);
  gemm_bf16<1, 16><<<dim3(1, M / 128), 256, 0, stream>>>(sb, wkb, Kp, M, 256, 256, 1.f);
  gemm_bf16<2, 16><<<dim3(1, M / 128), 256, 0, stream>>>(sb, wvb, Vsb, M, 256, 256,
                                                         1.0f / (float)SEQ);
  kv_reduce<<<BSZ * NHEAD, 256, 0, stream>>>(Kp, Vsb, KVt, Ksum, SEQ);
  attend<<<dim3(SEQ / 128, BSZ * NHEAD), 256, 0, stream>>>(Qp, KVt, Ksum, Att, SEQ, SEQ);

  // x1 = x + LN(Att @ wmerge^T)
  gemm_ln<<<M / 16, 256, 0, stream>>>(Att, wmb, x, g1, b1, x1, x1b, 256);
  // h = relu(x1 @ w1^T)
  gemm_bf16<3, 16><<<dim3(2, M / 128), 256, 0, stream>>>(x1b, w1b, Hb, M, 512, 256, 1.f);
  // out = x1 + LN(h @ w2^T)
  gemm_ln<<<M / 16, 256, 0, stream>>>(Hb, w2b, x1, g2, b2, (float*)d_out,
                                      (bfbits*)nullptr, 512);
}
